// Attention_53747220742271
// MI455X (gfx1250) — matrix operand order repaired and checked by reading, not compiled, not run on hardware
//
#include <hip/hip_runtime.h>
#include <hip/hip_bf16.h>

typedef __attribute__((ext_vector_type(16))) __bf16 v16bf;
typedef __attribute__((ext_vector_type(8)))  __bf16 v8bf;
typedef __attribute__((ext_vector_type(4)))  __bf16 v4bf;
typedef __attribute__((ext_vector_type(8)))  float  v8f;

#define LSEQ 4096
#define DDIM 1024
#define LDST 40   // LDS row stride in bf16 elems (80B: 16B-aligned, bank-spread)

__device__ inline v16bf concat8(v8bf lo, v8bf hi) {
  return __builtin_shufflevector(lo, hi, 0, 1, 2, 3, 4, 5, 6, 7,
                                 8, 9, 10, 11, 12, 13, 14, 15);
}

// A fragment (16x32 bf16) from [rows][LDST] tile, K-contiguous rows.
__device__ inline v16bf load_frag_A(const __bf16* s) {
  int lane = threadIdx.x & 31;
  const __bf16* p = s + (lane & 15) * LDST + ((lane >> 4) << 3);
  v8bf lo = *(const v8bf*)p;         // k = koff .. koff+7
  v8bf hi = *(const v8bf*)(p + 16);  // k = koff+16 .. koff+23
  return concat8(lo, hi);
}

// B fragment (32x16 bf16) from N-major [n][LDST] tile (K-contiguous per n).
__device__ inline v16bf load_frag_B(const __bf16* s, int coloff) {
  int lane = threadIdx.x & 31;
  const __bf16* p = s + (coloff + (lane & 15)) * LDST + ((lane >> 4) << 3);
  v8bf lo = *(const v8bf*)p;
  v8bf hi = *(const v8bf*)(p + 16);
  return concat8(lo, hi);
}

__device__ inline v4bf cvt4(const float4 f) {
  v4bf b;
  b[0] = (__bf16)f.x; b[1] = (__bf16)f.y;
  b[2] = (__bf16)f.z; b[3] = (__bf16)f.w;
  return b;
}

#define WMMA_BF16(A, B, C) \
  __builtin_amdgcn_wmma_f32_16x16x32_bf16(false, (A), false, (B), (short)0, (C), false, false)

// ---------------------------------------------------------------------------
// Kernel 1: out(bf16) = A(f32) @ W(f32) + bias.  Tile 128x64, BK=32, 4 waves,
// double-buffered LDS software pipeline.
// ---------------------------------------------------------------------------
__global__ __launch_bounds__(128) void proj_gemm_kernel(
    const float* __restrict__ A, const float* __restrict__ W,
    const float* __restrict__ bias, __bf16* __restrict__ out,
    int N, int K) {
  __shared__ __bf16 sA[2][128 * LDST];
  __shared__ __bf16 sB[2][64 * LDST];
  const int r0 = blockIdx.y * 128;
  const int c0 = blockIdx.x * 64;
  const int tid = threadIdx.x;
  const int lane = tid & 31, wave = tid >> 5;

  // per-thread staging coordinates (constant across chunks)
  const int ra = tid >> 3, ca = (tid & 7) * 4;    // A: 128 rows x 32k, 8 float4
  const int kbr = tid >> 4, cbn = (tid & 15) * 4; // B: 32k x 64n, 4 float4

  float4 fA[8];
  float4 fB[4];

  auto loadRegs = [&](int k0) {
#pragma unroll
    for (int u = 0; u < 8; ++u)
      fA[u] = *(const float4*)&A[(size_t)(r0 + ra + u * 16) * K + k0 + ca];
#pragma unroll
    for (int u = 0; u < 4; ++u)
      fB[u] = *(const float4*)&W[(size_t)(k0 + kbr + u * 8) * N + c0 + cbn];
  };
  auto storeRegs = [&](int p) {
#pragma unroll
    for (int u = 0; u < 8; ++u)
      *(v4bf*)&sA[p][(ra + u * 16) * LDST + ca] = cvt4(fA[u]);
#pragma unroll
    for (int u = 0; u < 4; ++u) {
      int k = kbr + u * 8;
      sB[p][(cbn + 0) * LDST + k] = (__bf16)fB[u].x;
      sB[p][(cbn + 1) * LDST + k] = (__bf16)fB[u].y;
      sB[p][(cbn + 2) * LDST + k] = (__bf16)fB[u].z;
      sB[p][(cbn + 3) * LDST + k] = (__bf16)fB[u].w;
    }
  };

  v8f acc[4][2] = {};
  const int nch = K / 32;
  loadRegs(0);
  storeRegs(0);
  int p = 0;
  for (int ch = 0; ch < nch; ++ch) {
    __syncthreads();                              // buf[p] staged
    if (ch + 1 < nch) loadRegs((ch + 1) * 32);    // issue global loads (no wait)
    v16bf a0 = load_frag_A(&sA[p][(wave * 32) * LDST]);
    v16bf a1 = load_frag_A(&sA[p][(wave * 32 + 16) * LDST]);
#pragma unroll
    for (int t = 0; t < 4; ++t) {
      v16bf bf = load_frag_B(&sB[p][0], t * 16);
      acc[t][0] = WMMA_BF16(a0, bf, acc[t][0]);
      acc[t][1] = WMMA_BF16(a1, bf, acc[t][1]);
    }
    if (ch + 1 < nch) storeRegs(p ^ 1);           // convert+store under WMMA shadow
    p ^= 1;
  }

  const int rb = r0 + wave * 32 + ((lane >> 4) ? 8 : 0);
  const int cb = lane & 15;
#pragma unroll
  for (int t = 0; t < 4; ++t) {
    int col = c0 + t * 16 + cb;
    float bv = bias[col];
#pragma unroll
    for (int h = 0; h < 2; ++h)
#pragma unroll
      for (int j = 0; j < 8; ++j)
        out[(size_t)(rb + h * 16 + j) * N + col] = (__bf16)(acc[t][h][j] + bv);
  }
}

// ---------------------------------------------------------------------------
// Kernel 2: S = (Q @ K^T)/32, masked (col<row -> -1e30). Tile 128x64.
// Fully-masked tiles (c0+64 <= r0) are filled without GEMM.
// ---------------------------------------------------------------------------
__global__ __launch_bounds__(128) void scores_kernel(
    const __bf16* __restrict__ Q, const __bf16* __restrict__ Kk,
    float* __restrict__ S) {
  const int r0 = blockIdx.y * 128;
  const int c0 = blockIdx.x * 64;
  const int tid = threadIdx.x;

  if (c0 + 64 <= r0) {  // entirely strictly-lower: fill -1e30, skip GEMM
    const float4 fill = make_float4(-1.0e30f, -1.0e30f, -1.0e30f, -1.0e30f);
#pragma unroll
    for (int u = 0; u < 16; ++u) {
      int r = (tid >> 4) + u * 8, c = (tid & 15) * 4;
      *(float4*)&S[(size_t)(r0 + r) * LSEQ + c0 + c] = fill;
    }
    return;
  }

  __shared__ __bf16 sQ[2][128 * LDST];
  __shared__ __bf16 sK[2][64 * LDST];
  const int lane = tid & 31, wave = tid >> 5;

  const int rq = tid >> 2, cq = (tid & 3) * 8;  // Q: 128 x 32, 4 v8bf per thread
  v8bf fQ[4];
  v8bf fK[2];

  auto loadRegs = [&](int k0) {
#pragma unroll
    for (int u = 0; u < 4; ++u)
      fQ[u] = *(const v8bf*)&Q[(size_t)(r0 + rq + u * 32) * DDIM + k0 + cq];
#pragma unroll
    for (int u = 0; u < 2; ++u)
      fK[u] = *(const v8bf*)&Kk[(size_t)(c0 + rq + u * 32) * DDIM + k0 + cq];
  };
  auto storeRegs = [&](int p) {
#pragma unroll
    for (int u = 0; u < 4; ++u)
      *(v8bf*)&sQ[p][(rq + u * 32) * LDST + cq] = fQ[u];
#pragma unroll
    for (int u = 0; u < 2; ++u)
      *(v8bf*)&sK[p][(rq + u * 32) * LDST + cq] = fK[u];
  };

  v8f acc[4][2] = {};
  const int nch = DDIM / 32;
  loadRegs(0);
  storeRegs(0);
  int p = 0;
  for (int ch = 0; ch < nch; ++ch) {
    __syncthreads();
    if (ch + 1 < nch) loadRegs((ch + 1) * 32);
    v16bf a0 = load_frag_A(&sQ[p][(wave * 32) * LDST]);
    v16bf a1 = load_frag_A(&sQ[p][(wave * 32 + 16) * LDST]);
#pragma unroll
    for (int t = 0; t < 4; ++t) {
      v16bf bf = load_frag_B(&sK[p][0], t * 16);
      acc[t][0] = WMMA_BF16(a0, bf, acc[t][0]);
      acc[t][1] = WMMA_BF16(a1, bf, acc[t][1]);
    }
    if (ch + 1 < nch) storeRegs(p ^ 1);
    p ^= 1;
  }

  const float scale = 0.03125f;  // 1/sqrt(1024)
  const int rb = r0 + wave * 32 + ((lane >> 4) ? 8 : 0);
  const int cb = lane & 15;
#pragma unroll
  for (int t = 0; t < 4; ++t) {
    int col = c0 + t * 16 + cb;
#pragma unroll
    for (int h = 0; h < 2; ++h)
#pragma unroll
      for (int j = 0; j < 8; ++j) {
        int row = rb + h * 16 + j;
        float v = acc[t][h][j] * scale;
        if (col < row) v = -1.0e30f;
        S[(size_t)row * LSEQ + col] = v;
      }
  }
}

// ---------------------------------------------------------------------------
// Kernel 3: in-place row softmax over S. One 256-thread block per row.
// ---------------------------------------------------------------------------
__global__ __launch_bounds__(256) void softmax_kernel(float* __restrict__ S) {
  const int row = blockIdx.x;
  const int tid = threadIdx.x;
  const int lane = tid & 31, wave = tid >> 5;
  float* rp = S + (size_t)row * LSEQ;

  float v[16];
  float m = -3.402823466e38f;
#pragma unroll
  for (int i = 0; i < 16; ++i) {
    v[i] = rp[tid + i * 256];
    m = fmaxf(m, v[i]);
  }
#pragma unroll
  for (int off = 16; off > 0; off >>= 1) m = fmaxf(m, __shfl_xor(m, off, 32));

  __shared__ float red[8];
  if (lane == 0) red[wave] = m;
  __syncthreads();
  float mt = red[0];
#pragma unroll
  for (int w = 1; w < 8; ++w) mt = fmaxf(mt, red[w]);
  m = mt;

  float s = 0.f;
#pragma unroll
  for (int i = 0; i < 16; ++i) {
    v[i] = __expf(v[i] - m);  // masked entries underflow to 0
    s += v[i];
  }
#pragma unroll
  for (int off = 16; off > 0; off >>= 1) s += __shfl_xor(s, off, 32);
  __syncthreads();
  if (lane == 0) red[wave] = s;
  __syncthreads();
  float st = 0.f;
#pragma unroll
  for (int w = 0; w < 8; ++w) st += red[w];

  const float inv = 1.0f / st;
#pragma unroll
  for (int i = 0; i < 16; ++i) rp[tid + i * 256] = v[i] * inv;
}

// ---------------------------------------------------------------------------
// Kernel 4: O(f32) = P(f32->bf16) @ V(bf16). Tile 128x64; K-loop starts at
// the diagonal block (P[row][k<r0]==0 for all rows in the tile).
// ---------------------------------------------------------------------------
__global__ __launch_bounds__(128) void pv_kernel(
    const float* __restrict__ P, const __bf16* __restrict__ V,
    float* __restrict__ O) {
  const int r0 = blockIdx.y * 128;
  const int c0 = blockIdx.x * 64;
  const int tid = threadIdx.x;
  const int lane = tid & 31, wave = tid >> 5;

  __shared__ __bf16 sP[2][128 * LDST];
  __shared__ __bf16 sV[2][64 * LDST];

  const int ra = tid >> 3, ca = (tid & 7) * 4;   // P: 128 x 32, 8 float4
  const int kv = tid >> 3, cv = (tid & 7) * 8;   // V: 32k x 64n, 2 v8bf

  float4 fP[8];
  v8bf fV[2];

  auto loadRegs = [&](int k0) {
#pragma unroll
    for (int u = 0; u < 8; ++u)
      fP[u] = *(const float4*)&P[(size_t)(r0 + ra + u * 16) * LSEQ + k0 + ca];
#pragma unroll
    for (int u = 0; u < 2; ++u)
      fV[u] = *(const v8bf*)&V[(size_t)(k0 + kv + u * 16) * DDIM + c0 + cv];
  };
  auto storeRegs = [&](int p) {
#pragma unroll
    for (int u = 0; u < 8; ++u)
      *(v4bf*)&sP[p][(ra + u * 16) * LDST + ca] = cvt4(fP[u]);
#pragma unroll
    for (int u = 0; u < 2; ++u) {
      int k = kv + u * 16;
#pragma unroll
      for (int t = 0; t < 8; ++t) sV[p][(cv + t) * LDST + k] = fV[u][t];
    }
  };

  v8f acc[4][2] = {};
  const int nch = (LSEQ - r0) / 32;
  loadRegs(r0);
  storeRegs(0);
  int p = 0;
  for (int ch = 0; ch < nch; ++ch) {
    __syncthreads();
    if (ch + 1 < nch) loadRegs(r0 + (ch + 1) * 32);
    v16bf a0 = load_frag_A(&sP[p][(wave * 32) * LDST]);
    v16bf a1 = load_frag_A(&sP[p][(wave * 32 + 16) * LDST]);
#pragma unroll
    for (int t = 0; t < 4; ++t) {
      v16bf bf = load_frag_B(&sV[p][0], t * 16);
      acc[t][0] = WMMA_BF16(a0, bf, acc[t][0]);
      acc[t][1] = WMMA_BF16(a1, bf, acc[t][1]);
    }
    if (ch + 1 < nch) storeRegs(p ^ 1);
    p ^= 1;
  }

  const int rb = r0 + wave * 32 + ((lane >> 4) ? 8 : 0);
  const int cb = lane & 15;
#pragma unroll
  for (int t = 0; t < 4; ++t) {
    int col = c0 + t * 16 + cb;
#pragma unroll
    for (int h = 0; h < 2; ++h)
#pragma unroll
      for (int j = 0; j < 8; ++j)
        O[(size_t)(rb + h * 16 + j) * DDIM + col] = acc[t][h][j];
  }
}

// ---------------------------------------------------------------------------
extern "C" void kernel_launch(void* const* d_in, const int* in_sizes, int n_in,
                              void* d_out, int out_size, void* d_ws, size_t ws_size,
                              hipStream_t stream) {
  const float* x  = (const float*)d_in[0];
  const float* z  = (const float*)d_in[1];
  const float* Wq = (const float*)d_in[2];
  const float* bq = (const float*)d_in[3];
  const float* Wk = (const float*)d_in[4];
  const float* bk = (const float*)d_in[5];
  const float* Wv = (const float*)d_in[6];
  const float* bv = (const float*)d_in[7];
  float* out = (float*)d_out;

  char* ws = (char*)d_ws;
  __bf16* Qb = (__bf16*)(ws);                        // 8 MB
  __bf16* Kb = (__bf16*)(ws + ((size_t)8 << 20));    // 8 MB
  __bf16* Vb = (__bf16*)(ws + ((size_t)16 << 20));   // 8 MB
  float*  S  = (float*)(ws + ((size_t)24 << 20));    // 64 MB (L2-resident)

  dim3 gProj(DDIM / 64, LSEQ / 128);  // (16, 32)
  proj_gemm_kernel<<<gProj, 128, 0, stream>>>(x, Wq, bq, Qb, DDIM, DDIM);
  proj_gemm_kernel<<<gProj, 128, 0, stream>>>(z, Wk, bk, Kb, DDIM, DDIM);
  proj_gemm_kernel<<<gProj, 128, 0, stream>>>(z, Wv, bv, Vb, DDIM, DDIM);

  dim3 gS(LSEQ / 64, LSEQ / 128);     // (64, 32)
  scores_kernel<<<gS, 128, 0, stream>>>(Qb, Kb, S);

  softmax_kernel<<<LSEQ, 256, 0, stream>>>(S);

  dim3 gO(DDIM / 64, LSEQ / 128);     // (16, 32)
  pv_kernel<<<gO, 128, 0, stream>>>(S, Vb, out);
}
